// GRUNet_81157702025423
// MI455X (gfx1250) — matrix operand order repaired and checked by reading, not compiled, not run on hardware
//
#include <hip/hip_runtime.h>
#include <hip/hip_bf16.h>

#define HID    64
#define G3     192
#define BATCH  2048
#define TSTEPS 1024
#define NBT    2          // batch tiles (of 16) per workgroup, shared by all waves
#define NW     4          // waves per workgroup (wave j owns gate/hidden block j)

typedef __bf16 bf16;
typedef __attribute__((ext_vector_type(16))) __bf16 v16bf;
typedef __attribute__((ext_vector_type(8)))  __bf16 v8bf;
typedef __attribute__((ext_vector_type(8)))  float  v8f;

static __device__ __forceinline__ v8f wmma_bf16(v16bf a, v16bf b, v8f c) {
  // D = A(16x32) * B(32x16) + C, f32 accumulate
  return __builtin_amdgcn_wmma_f32_16x16x32_bf16(false, a, false, b, (short)0, c,
                                                 false, false);
}

// --- fast elementwise transcendentals (keep the serial per-step chain short)
static __device__ __forceinline__ float fast_rcp(float x) {
  return __builtin_amdgcn_rcpf(x);   // v_rcp_f32
}

#if __has_builtin(__builtin_amdgcn_tanhf)
#define GRU_HAVE_TANH 1
static __device__ __forceinline__ float tanh_fast(float x) {
  return __builtin_amdgcn_tanhf(x);  // v_tanh_f32 (CDNA5 TRANS op)
}
#elif __has_builtin(__builtin_amdgcn_tanh_f32)
#define GRU_HAVE_TANH 1
static __device__ __forceinline__ float tanh_fast(float x) {
  return __builtin_amdgcn_tanh_f32(x);
}
#else
static __device__ __forceinline__ float tanh_fast(float x) {
  return 2.0f * fast_rcp(1.0f + __expf(-2.0f * x)) - 1.0f;  // v_exp + v_rcp
}
#endif

static __device__ __forceinline__ float sigf(float x) {
#ifdef GRU_HAVE_TANH
  return __builtin_fmaf(0.5f, tanh_fast(0.5f * x), 0.5f);   // 1 trans op
#else
  return fast_rcp(1.0f + __expf(-x));
#endif
}

// A fragment (16x32 bf16): weights in LDS, row-major [gate g][k].
// lane: M = ln; elems 0..7 -> K = kc*32 + 8*lh + e; elems 8..15 -> +16.
static __device__ __forceinline__ v16bf load_afrag(const bf16* w, int gt, int kc,
                                                   int ln, int lh) {
  const bf16* p = w + (gt * 16 + ln) * HID + kc * 32 + lh * 8;
  v8bf lo = *(const v8bf*)p;
  v8bf hi = *(const v8bf*)(p + 16);
  return __builtin_shufflevector(lo, hi, 0, 1, 2, 3, 4, 5, 6, 7,
                                 8, 9, 10, 11, 12, 13, 14, 15);
}

// B fragment (32x16 bf16): h in LDS row-major [batch][hidden].
// lane: N = batch col = ln; elems 0..7 -> K = kc*32 + 8*lh + e; elems 8..15 -> +16.
static __device__ __forceinline__ v16bf load_bfrag(const bf16* h, int row, int kc,
                                                   int lh) {
  const bf16* p = h + row * HID + kc * 32 + lh * 8;
  v8bf lo = *(const v8bf*)p;
  v8bf hi = *(const v8bf*)(p + 16);
  return __builtin_shufflevector(lo, hi, 0, 1, 2, 3, 4, 5, 6, 7,
                                 8, 9, 10, 11, 12, 13, 14, 15);
}

// ---------------------------------------------------------------------------
// x [B,T] -> xT [T,B]  (coalesced LDS tile transpose)
// ---------------------------------------------------------------------------
__global__ void gru_xpose(const float* __restrict__ x, float* __restrict__ xT) {
  __shared__ float tile[32][33];
  const int tb = blockIdx.x * 32;  // t block
  const int bb = blockIdx.y * 32;  // batch block
  const int lx = threadIdx.x, ly = threadIdx.y;
#pragma unroll
  for (int i = 0; i < 32; i += 8)
    tile[ly + i][lx] = x[(size_t)(bb + ly + i) * TSTEPS + tb + lx];
  __syncthreads();
#pragma unroll
  for (int i = 0; i < 32; i += 8)
    xT[(size_t)(tb + ly + i) * BATCH + bb + lx] = tile[lx][ly + i];
}

// ---------------------------------------------------------------------------
// Fused 2-layer GRU scan + FC head. 1 workgroup = 4 waves x 32 batch cols.
// ---------------------------------------------------------------------------
__global__ void __launch_bounds__(128, 1) gru_fused(
    const float* __restrict__ xT,
    const float* __restrict__ w_ih0, const float* __restrict__ w_hh0,
    const float* __restrict__ b_ih0, const float* __restrict__ b_hh0,
    const float* __restrict__ w_ih1, const float* __restrict__ w_hh1,
    const float* __restrict__ b_ih1, const float* __restrict__ b_hh1,
    const float* __restrict__ w_fc, const float* __restrict__ b_fc,
    float* __restrict__ out) {
  __shared__ struct __align__(64) {
    bf16  w[3][G3 * HID];           // bf16 weights: w_hh0, w_ih1, w_hh1
    bf16  h0[2][NBT * 16][HID];     // layer-0 hidden state (ping-pong)
    bf16  h1[2][NBT * 16][HID];     // layer-1 hidden state (ping-pong)
    float oacc[NBT * 16];
  } sm;

  const int tid = threadIdx.x;
  const int wv = tid >> 5, lane = tid & 31;
  const int ln = lane & 15, lh = lane >> 4;
  const int gtR = wv, gtZ = 4 + wv, gtN = 8 + wv;   // gate tiles owned by wave
  const int wgBase = blockIdx.x * (NBT * 16);

  // ---- preamble: weights -> LDS bf16, zero state buffers -------------------
  for (int i = tid * 4; i < 3 * G3 * HID; i += 128 * 4) {
    const float* src = (i < G3 * HID)     ? (w_hh0 + i)
                       : (i < 2 * G3 * HID) ? (w_ih1 + i - G3 * HID)
                                            : (w_hh1 + i - 2 * G3 * HID);
    bf16* dst = &sm.w[0][0] + i;
#pragma unroll
    for (int u = 0; u < 4; ++u) dst[u] = (bf16)src[u];
  }
  {
    bf16* hz0 = &sm.h0[0][0][0];
    bf16* hz1 = &sm.h1[0][0][0];
    for (int i = tid; i < 2 * NBT * 16 * HID; i += 128) {
      hz0[i] = (bf16)0.0f;
      hz1[i] = (bf16)0.0f;
    }
    if (tid < NBT * 16) sm.oacc[tid] = 0.0f;
  }

  // ---- per-lane loop-invariant scalars (C/D elem r <-> gate row +8*lh) -----
  float wi0r[8], wi0z[8], wi0n[8];
  float bs0r[8], bs0z[8], bi0n[8], bh0n[8];
  float bs1r[8], bs1z[8], bi1n[8], bh1n[8];
  float wfc[8];
#pragma unroll
  for (int r = 0; r < 8; ++r) {
    const int gR = 16 * gtR + r + 8 * lh;
    const int gZ = 16 * gtZ + r + 8 * lh;
    const int gN = 16 * gtN + r + 8 * lh;
    wi0r[r] = w_ih0[gR]; wi0z[r] = w_ih0[gZ]; wi0n[r] = w_ih0[gN];
    bs0r[r] = b_ih0[gR] + b_hh0[gR];
    bs0z[r] = b_ih0[gZ] + b_hh0[gZ];
    bi0n[r] = b_ih0[gN]; bh0n[r] = b_hh0[gN];
    bs1r[r] = b_ih1[gR] + b_hh1[gR];
    bs1z[r] = b_ih1[gZ] + b_hh1[gZ];
    bi1n[r] = b_ih1[gN]; bh1n[r] = b_hh1[gN];
    wfc[r] = w_fc[16 * wv + r + 8 * lh];
  }
  float h0p[NBT][8] = {};   // previous hidden state, f32, C/D layout
  float h1p[NBT][8] = {};
  __syncthreads();

  const bf16* W0  = &sm.w[0][0];
  const bf16* Wi1 = &sm.w[1][0];
  const bf16* Wh1 = &sm.w[2][0];

  // ---- recurrent scan ------------------------------------------------------
  for (int t = 0; t < TSTEPS; ++t) {
    const int cur = t & 1, prv = cur ^ 1;

    // ===== layer 0: gh0 = w_hh0 @ h0^T ; gx0 is rank-1 (I==1), VALU-side ====
    v16bf bh0[NBT][2];
    float xv[NBT];
#pragma unroll
    for (int bt = 0; bt < NBT; ++bt) {
      xv[bt] = xT[(size_t)t * BATCH + wgBase + bt * 16 + ln];
      bh0[bt][0] = load_bfrag(&sm.h0[prv][0][0], bt * 16 + ln, 0, lh);
      bh0[bt][1] = load_bfrag(&sm.h0[prv][0][0], bt * 16 + ln, 1, lh);
    }
    v8f aR[NBT], aZ[NBT], aN[NBT];
    float xn0[NBT][8];
#pragma unroll
    for (int bt = 0; bt < NBT; ++bt)
#pragma unroll
      for (int r = 0; r < 8; ++r) {
        aR[bt][r] = __builtin_fmaf(xv[bt], wi0r[r], bs0r[r]);  // xr (+biases)
        aZ[bt][r] = __builtin_fmaf(xv[bt], wi0z[r], bs0z[r]);
        aN[bt][r] = bh0n[r];                                   // hn acc (b_hh)
        xn0[bt][r] = __builtin_fmaf(xv[bt], wi0n[r], bi0n[r]); // xn separate
      }
#pragma unroll
    for (int kc = 0; kc < 2; ++kc) {
      v16bf fr = load_afrag(W0, gtR, kc, ln, lh);
#pragma unroll
      for (int bt = 0; bt < NBT; ++bt) aR[bt] = wmma_bf16(fr, bh0[bt][kc], aR[bt]);
      v16bf fz = load_afrag(W0, gtZ, kc, ln, lh);
#pragma unroll
      for (int bt = 0; bt < NBT; ++bt) aZ[bt] = wmma_bf16(fz, bh0[bt][kc], aZ[bt]);
      v16bf fn = load_afrag(W0, gtN, kc, ln, lh);
#pragma unroll
      for (int bt = 0; bt < NBT; ++bt) aN[bt] = wmma_bf16(fn, bh0[bt][kc], aN[bt]);
    }
#pragma unroll
    for (int bt = 0; bt < NBT; ++bt) {
      v8bf pk;
#pragma unroll
      for (int r = 0; r < 8; ++r) {
        const float rg = sigf(aR[bt][r]);
        const float zg = sigf(aZ[bt][r]);
        const float ng = tanh_fast(__builtin_fmaf(rg, aN[bt][r], xn0[bt][r]));
        const float h = __builtin_fmaf(zg, h0p[bt][r] - ng, ng); // (1-z)n + z h
        h0p[bt][r] = h;
        pk[r] = (bf16)h;
      }
      // lane holds 8 consecutive hidden values of one batch col -> 1x b128
      *(v8bf*)&sm.h0[cur][bt * 16 + ln][16 * wv + 8 * lh] = pk;
    }
    __syncthreads();  // all waves' h0_new visible for layer-1 K=64 reads

    // ===== layer 1: g1 = w_ih1 @ h0new^T + w_hh1 @ h1^T =====================
    v16bf bx[NBT][2], bh1[NBT][2];
#pragma unroll
    for (int bt = 0; bt < NBT; ++bt) {
      bx[bt][0]  = load_bfrag(&sm.h0[cur][0][0], bt * 16 + ln, 0, lh);
      bx[bt][1]  = load_bfrag(&sm.h0[cur][0][0], bt * 16 + ln, 1, lh);
      bh1[bt][0] = load_bfrag(&sm.h1[prv][0][0], bt * 16 + ln, 0, lh);
      bh1[bt][1] = load_bfrag(&sm.h1[prv][0][0], bt * 16 + ln, 1, lh);
    }
    v8f r1[NBT], z1[NBT], x1[NBT], n1[NBT];
#pragma unroll
    for (int bt = 0; bt < NBT; ++bt)
#pragma unroll
      for (int r = 0; r < 8; ++r) {
        r1[bt][r] = bs1r[r];
        z1[bt][r] = bs1z[r];
        x1[bt][r] = bi1n[r];
        n1[bt][r] = bh1n[r];
      }
#pragma unroll
    for (int kc = 0; kc < 2; ++kc) {
      v16bf fxr = load_afrag(Wi1, gtR, kc, ln, lh);
#pragma unroll
      for (int bt = 0; bt < NBT; ++bt) r1[bt] = wmma_bf16(fxr, bx[bt][kc], r1[bt]);
      v16bf fhr = load_afrag(Wh1, gtR, kc, ln, lh);
#pragma unroll
      for (int bt = 0; bt < NBT; ++bt) r1[bt] = wmma_bf16(fhr, bh1[bt][kc], r1[bt]);
      v16bf fxz = load_afrag(Wi1, gtZ, kc, ln, lh);
#pragma unroll
      for (int bt = 0; bt < NBT; ++bt) z1[bt] = wmma_bf16(fxz, bx[bt][kc], z1[bt]);
      v16bf fhz = load_afrag(Wh1, gtZ, kc, ln, lh);
#pragma unroll
      for (int bt = 0; bt < NBT; ++bt) z1[bt] = wmma_bf16(fhz, bh1[bt][kc], z1[bt]);
      v16bf fxn = load_afrag(Wi1, gtN, kc, ln, lh);
#pragma unroll
      for (int bt = 0; bt < NBT; ++bt) x1[bt] = wmma_bf16(fxn, bx[bt][kc], x1[bt]);
      v16bf fhn = load_afrag(Wh1, gtN, kc, ln, lh);
#pragma unroll
      for (int bt = 0; bt < NBT; ++bt) n1[bt] = wmma_bf16(fhn, bh1[bt][kc], n1[bt]);
    }
#pragma unroll
    for (int bt = 0; bt < NBT; ++bt) {
      v8bf pk;
#pragma unroll
      for (int r = 0; r < 8; ++r) {
        const float rg = sigf(r1[bt][r]);
        const float zg = sigf(z1[bt][r]);
        const float ng = tanh_fast(__builtin_fmaf(rg, n1[bt][r], x1[bt][r]));
        const float h = __builtin_fmaf(zg, h1p[bt][r] - ng, ng);
        h1p[bt][r] = h;
        pk[r] = (bf16)h;
      }
      *(v8bf*)&sm.h1[cur][bt * 16 + ln][16 * wv + 8 * lh] = pk;
    }
    __syncthreads();  // protects ping-pong WAR across waves
  }

  // ---- FC head: out[b] = h1_last[b,:] . w_fc + b_fc ------------------------
#pragma unroll
  for (int bt = 0; bt < NBT; ++bt) {
    float p = 0.0f;
#pragma unroll
    for (int r = 0; r < 8; ++r) p = __builtin_fmaf(h1p[bt][r], wfc[r], p);
    atomicAdd(&sm.oacc[bt * 16 + ln], p);  // ds_add_f32 reduce over hidden
  }
  __syncthreads();
  if (tid < NBT * 16) out[wgBase + tid] = sm.oacc[tid] + b_fc[0];
}

extern "C" void kernel_launch(void* const* d_in, const int* in_sizes, int n_in,
                              void* d_out, int out_size, void* d_ws, size_t ws_size,
                              hipStream_t stream) {
  (void)in_sizes; (void)n_in; (void)out_size; (void)ws_size;
  const float* x     = (const float*)d_in[0];
  const float* w_ih0 = (const float*)d_in[1];
  const float* w_hh0 = (const float*)d_in[2];
  const float* b_ih0 = (const float*)d_in[3];
  const float* b_hh0 = (const float*)d_in[4];
  const float* w_ih1 = (const float*)d_in[5];
  const float* w_hh1 = (const float*)d_in[6];
  const float* b_ih1 = (const float*)d_in[7];
  const float* b_hh1 = (const float*)d_in[8];
  const float* w_fc  = (const float*)d_in[9];
  const float* b_fc  = (const float*)d_in[10];
  float* xT = (float*)d_ws;  // [T, B] f32 = 8 MB

  gru_xpose<<<dim3(TSTEPS / 32, BATCH / 32), dim3(32, 8), 0, stream>>>(x, xT);
  gru_fused<<<dim3(BATCH / (NBT * 16)), dim3(128), 0, stream>>>(
      xT, w_ih0, w_hh0, b_ih0, b_hh0, w_ih1, w_hh1, b_ih1, b_hh1, w_fc, b_fc,
      (float*)d_out);
}
